// DeformableBlock_79637283602552
// MI455X (gfx1250) — matrix operand order repaired and checked by reading, not compiled, not run on hardware
//
#include <hip/hip_runtime.h>

typedef __attribute__((ext_vector_type(16))) _Float16 v16h;
typedef __attribute__((ext_vector_type(8)))  float    v8f;

#define BB     8
#define CIN    64
#define COUT   64
#define HH     128
#define WW     128
#define HWSZ   (HH*WW)            // 16384
#define KDIM   (9*CIN)            // 576
#define KSTEPS (KDIM/32)          // 18
#define NPIX   (BB*HWSZ)          // 131072
#define NTILES (NPIX/16)          // 8192
#define WAVES_PER_BLOCK 8
#define NT_PER_WAVE 4
#define NBLOCKS (NTILES/(WAVES_PER_BLOCK*NT_PER_WAVE))   // 256

__device__ __forceinline__ int iclamp(int v, int lo, int hi) {
    return v < lo ? lo : (v > hi ? hi : v);
}

// Map (mtile, kstep, lane, halfIdx) -> global weight row/K in this file's
// contiguous k order (B matches; not the ISA table order):
//   M = (lane & 15)           (both lane halves cover M=0..15)
//   v = h>>1 (VGPR), lo/hi = h&1
//   K = 32*kk + 16*(lane>=16) + h
__device__ __forceinline__ int afrag_K(int kk, int lane, int h) {
    return kk * 32 + ((lane >> 4) << 4) + h;
}

// ---------------------------------------------------------------------------
// Kernel 1: offset conv (3x3, 64->18, SAME, zero pad) via WMMA f16->f32.
// Weights pre-swizzled in LDS into A-fragment order: one aligned v16h load
// per fragment.  Output: off[b][ch][h][w] fp32, ch = 2*tap (+0 dy, +1 dx).
// ---------------------------------------------------------------------------
__global__ __launch_bounds__(256) void offset_conv_wmma(
    const float* __restrict__ x, const float* __restrict__ ow,
    const float* __restrict__ ob, float* __restrict__ off)
{
    // [mtile(2)][kstep(18)][lane(32)][half(16)] = 18432 halves = 36 KB
    __shared__ __align__(32) _Float16 sW[2 * KSTEPS * 32 * 16];
    for (int i = threadIdx.x; i < 2 * KSTEPS * 32 * 16; i += 256) {
        int h    = i & 15;
        int frag = i >> 4;
        int lane = frag & 31;
        int fk   = frag >> 5;
        int kk   = fk % KSTEPS;
        int m    = fk / KSTEPS;
        int M    = m * 16 + (lane & 15);          // padded output row (0..31)
        int K    = afrag_K(kk, lane, h);
        int tap  = K >> 6, c = K & 63;
        float wv = (M < 18) ? ow[(M * CIN + c) * 9 + tap] : 0.0f;
        sW[i] = (_Float16)wv;
    }
    __syncthreads();

    const int lane = threadIdx.x & 31;
    const int wave = threadIdx.x >> 5;
    const int hi   = lane >> 4;          // 0: lanes 0-15, 1: lanes 16-31
    const int ln   = lane & 15;

#pragma unroll 1
    for (int it = 0; it < NT_PER_WAVE; ++it) {
        int ntile = (blockIdx.x * WAVES_PER_BLOCK + wave) * NT_PER_WAVE + it;
        int g   = ntile * 16 + ln;       // flat pixel for this lane's N column
        int b   = g >> 14;
        int rem = g & (HWSZ - 1);
        int y   = rem >> 7;
        int xx  = rem & (WW - 1);

        v8f acc0 = {}; v8f acc1 = {};
#pragma unroll 1
        for (int kk = 0; kk < KSTEPS; ++kk) {
            int Kbase = kk * 32 + hi * 16;
            int tap = Kbase >> 6;
            int c0  = Kbase & 63;        // 16-wide K window stays in one tap
            int ki = tap / 3 - 1, kj = tap % 3 - 1;
            int ys = y + ki, xs = xx + kj;
            float msk = (ys >= 0 && ys < HH && xs >= 0 && xs < WW) ? 1.0f : 0.0f;
            int yc = iclamp(ys, 0, HH - 1), xc = iclamp(xs, 0, WW - 1);
            const float* xb = x + ((size_t)(b * CIN + c0)) * HWSZ + (yc * WW + xc);

            union { v16h v; _Float16 h[16]; } bf;
#pragma unroll
            for (int h = 0; h < 16; ++h)
                bf.h[h] = (_Float16)(msk * xb[h * HWSZ]);

            const v16h a0 = *(const v16h*)(sW + (((0 * KSTEPS + kk) * 32 + lane) << 4));
            const v16h a1 = *(const v16h*)(sW + (((1 * KSTEPS + kk) * 32 + lane) << 4));
            acc0 = __builtin_amdgcn_wmma_f32_16x16x32_f16(
                false, a0, false, bf.v, (short)0, acc0, false, false);
            acc1 = __builtin_amdgcn_wmma_f32_16x16x32_f16(
                false, a1, false, bf.v, (short)0, acc1, false, false);
        }
#pragma unroll
        for (int m = 0; m < 2; ++m) {
#pragma unroll
            for (int r = 0; r < 8; ++r) {
                int o = m * 16 + r + hi * 8;     // C/D layout: M = r + 8*hi
                if (o < 18) {
                    float vv = (m == 0 ? acc0[r] : acc1[r]) + ob[o];
                    off[((size_t)(b * 18 + o)) * HWSZ + rem] = vv;
                }
            }
        }
    }
}

// ---------------------------------------------------------------------------
// Kernel 2: bilinear sample (L2-resident x) fused with main conv GEMM
// (M=64, K=576, N=pixels) via WMMA f16->f32, + bias + ReLU.
// ---------------------------------------------------------------------------
__global__ __launch_bounds__(256) void deform_conv_wmma(
    const float* __restrict__ x, const float* __restrict__ dw,
    const float* __restrict__ db, const float* __restrict__ off,
    float* __restrict__ out)
{
    // [mtile(4)][kstep(18)][lane(32)][half(16)] = 36864 halves = 72 KB
    __shared__ __align__(32) _Float16 sW[4 * KSTEPS * 32 * 16];
    for (int i = threadIdx.x; i < 4 * KSTEPS * 32 * 16; i += 256) {
        int h    = i & 15;
        int frag = i >> 4;
        int lane = frag & 31;
        int fk   = frag >> 5;
        int kk   = fk % KSTEPS;
        int m    = fk / KSTEPS;
        int M    = m * 16 + (lane & 15);
        int K    = afrag_K(kk, lane, h);
        int tap  = K >> 6, c = K & 63;
        sW[i] = (_Float16)dw[(M * CIN + c) * 9 + tap];
    }
    __syncthreads();

    const int lane = threadIdx.x & 31;
    const int wave = threadIdx.x >> 5;
    const int hi   = lane >> 4;
    const int ln   = lane & 15;

#pragma unroll 1
    for (int it = 0; it < NT_PER_WAVE; ++it) {
        int ntile = (blockIdx.x * WAVES_PER_BLOCK + wave) * NT_PER_WAVE + it;
        int g   = ntile * 16 + ln;
        int b   = g >> 14;
        int rem = g & (HWSZ - 1);
        int y   = rem >> 7;
        int xx  = rem & (WW - 1);

        v8f acc[4] = {v8f{}, v8f{}, v8f{}, v8f{}};
#pragma unroll 1
        for (int kk = 0; kk < KSTEPS; ++kk) {
            int Kbase = kk * 32 + hi * 16;
            int tap = Kbase >> 6;
            int c0  = Kbase & 63;
            int ki = tap / 3 - 1, kj = tap % 3 - 1;

            // Deformed sample position (shared across the 16 cin of this step)
            float dy = off[((size_t)(b * 18 + 2 * tap)) * HWSZ + rem];
            float dx = off[((size_t)(b * 18 + 2 * tap + 1)) * HWSZ + rem];
            float py = (float)(y + ki) + dy;
            float px = (float)(xx + kj) + dx;
            float y0f = floorf(py), x0f = floorf(px);
            float wy = py - y0f, wx = px - x0f;
            int y0 = (int)y0f, x0 = (int)x0f;
            int y1 = y0 + 1,  x1 = x0 + 1;
            float vy0 = (y0 >= 0 && y0 < HH) ? 1.0f : 0.0f;
            float vy1 = (y1 >= 0 && y1 < HH) ? 1.0f : 0.0f;
            float vx0 = (x0 >= 0 && x0 < WW) ? 1.0f : 0.0f;
            float vx1 = (x1 >= 0 && x1 < WW) ? 1.0f : 0.0f;
            float w00 = (1.0f - wy) * (1.0f - wx) * vy0 * vx0;
            float w01 = (1.0f - wy) * wx * vy0 * vx1;
            float w10 = wy * (1.0f - wx) * vy1 * vx0;
            float w11 = wy * wx * vy1 * vx1;
            int yc0 = iclamp(y0, 0, HH - 1), yc1 = iclamp(y1, 0, HH - 1);
            int xc0 = iclamp(x0, 0, WW - 1), xc1 = iclamp(x1, 0, WW - 1);
            int s00 = yc0 * WW + xc0, s01 = yc0 * WW + xc1;
            int s10 = yc1 * WW + xc0, s11 = yc1 * WW + xc1;
            const float* xb = x + ((size_t)(b * CIN + c0)) * HWSZ;

            union { v16h v; _Float16 h[16]; } bf;
#pragma unroll
            for (int h = 0; h < 16; ++h) {
                int co = h * HWSZ;
                float sv = w00 * xb[co + s00] + w01 * xb[co + s01]
                         + w10 * xb[co + s10] + w11 * xb[co + s11];
                bf.h[h] = (_Float16)sv;
            }

#pragma unroll
            for (int m = 0; m < 4; ++m) {
                const v16h a = *(const v16h*)(sW + (((m * KSTEPS + kk) * 32 + lane) << 4));
                acc[m] = __builtin_amdgcn_wmma_f32_16x16x32_f16(
                    false, a, false, bf.v, (short)0, acc[m], false, false);
            }
        }
#pragma unroll
        for (int m = 0; m < 4; ++m) {
#pragma unroll
            for (int r = 0; r < 8; ++r) {
                int o = m * 16 + r + hi * 8;
                float vv = acc[m][r] + db[o];
                vv = fmaxf(vv, 0.0f);
                out[((size_t)(b * COUT + o)) * HWSZ + rem] = vv;
            }
        }
    }
}

extern "C" void kernel_launch(void* const* d_in, const int* in_sizes, int n_in,
                              void* d_out, int out_size, void* d_ws, size_t ws_size,
                              hipStream_t stream) {
    const float* x  = (const float*)d_in[0];   // (8,64,128,128)
    const float* ow = (const float*)d_in[1];   // (18,64,3,3)
    const float* ob = (const float*)d_in[2];   // (18,)
    const float* dw = (const float*)d_in[3];   // (64,64,3,3)
    const float* db = (const float*)d_in[4];   // (64,)
    float* offb = (float*)d_ws;                // 8*18*16384*4 = 9.4 MB
    float* out  = (float*)d_out;               // (8,64,128,128)

    offset_conv_wmma<<<NBLOCKS, 256, 0, stream>>>(x, ow, ob, offb);
    deform_conv_wmma<<<NBLOCKS, 256, 0, stream>>>(x, dw, db, offb, out);
}
